// BertAttention_28896539967576
// MI455X (gfx1250) — matrix operand order repaired and checked by reading, not compiled, not run on hardware
//
#include <hip/hip_runtime.h>
#include <hip/hip_bf16.h>

#define H   1024
#define NH  16
#define HD  64
#define S   2048
#define BB  2
#define NTOK (BB * S)   // 4096

typedef __bf16 bf16_t;
typedef __attribute__((ext_vector_type(16))) __bf16 v16bf;
typedef __attribute__((ext_vector_type(8)))  __bf16 v8bf;
typedef __attribute__((ext_vector_type(8)))  float  v8f;

// ---------------------------------------------------------------------------
// WMMA helpers (ISA 7.12.2 layouts, wave32)
// ---------------------------------------------------------------------------
static __device__ __forceinline__ v16bf bf_combine(v8bf lo, v8bf hi) {
    v16bf r;
#pragma unroll
    for (int i = 0; i < 8; ++i) { r[i] = lo[i]; r[i + 8] = hi[i]; }
    return r;
}

// A-matrix 16x32 bf16 tile: lane(l15,half) holds row l15,
// K contiguous {k+half*16 .. +15} (matches B operand order)
static __device__ __forceinline__ v16bf load_a_tile(const bf16_t* rowp, int k, int half) {
    v8bf lo = *(const v8bf*)(rowp + k + half * 16);
    v8bf hi = *(const v8bf*)(rowp + k + half * 16 + 8);
    return bf_combine(lo, hi);
}

static __device__ __forceinline__ v8f wmma_bf16(v16bf a, v16bf b, v8f c) {
    // (neg_a, A, neg_b, B, c_mod, C, reuse_a, reuse_b)
    return __builtin_amdgcn_wmma_f32_16x16x32_bf16(false, a, false, b, (short)0, c,
                                                   false, false);
}

// 64x64 output tile per wave: 16 WMMAs per 8 loads -> ~32 FLOP/byte from L2.
// A row-major [M][K], Bt row-major [N][K] (i.e. B^T).
static __device__ __forceinline__ void wave_gemm_64x64(
    const bf16_t* __restrict__ A, int lda,
    const bf16_t* __restrict__ Bt, int ldb,
    int K, int l15, int half, v8f c[4][4]) {
    const bf16_t* ar[4];
    const bf16_t* br[4];
#pragma unroll
    for (int i = 0; i < 4; ++i) {
        ar[i] = A  + (size_t)(l15 + 16 * i) * lda;
        br[i] = Bt + (size_t)(l15 + 16 * i) * ldb + half * 16;
    }
#pragma unroll 1
    for (int k = 0; k < K; k += 32) {
        v16bf a[4], b[4];
#pragma unroll
        for (int i = 0; i < 4; ++i) a[i] = load_a_tile(ar[i], k, half);
#pragma unroll
        for (int j = 0; j < 4; ++j) b[j] = *(const v16bf*)(br[j] + k);  // 32B/lane
#pragma unroll
        for (int i = 0; i < 4; ++i)
#pragma unroll
            for (int j = 0; j < 4; ++j)
                c[i][j] = wmma_bf16(a[i], b[j], c[i][j]);
    }
}

// ---------------------------------------------------------------------------
// Precision conversion / weight transpose
// ---------------------------------------------------------------------------
__global__ __launch_bounds__(256) void f32_to_bf16_kernel(
    const float* __restrict__ x, bf16_t* __restrict__ y, int n) {
    int i = blockIdx.x * 256 + threadIdx.x;
    if (i < n) y[i] = (bf16_t)x[i];
}

// WT[n][k] = (bf16)W[k][n]  (so GEMM B-tiles are contiguous per lane)
__global__ __launch_bounds__(256) void transpose_bf16_kernel(
    const float* __restrict__ W, bf16_t* __restrict__ WT) {
    int n = blockIdx.x * 16 + (threadIdx.x & 15);
    int k = blockIdx.y * 16 + (threadIdx.x >> 4);
    WT[(size_t)n * H + k] = (bf16_t)W[(size_t)k * H + n];
}

// ---------------------------------------------------------------------------
// Fused QKV projection.  grid (32, 8, 3), block 128 (4 waves, 2x2 of 64x64).
// z=0 -> Q [bh][s][d] bf16, z=1 -> K [bh][s][d] bf16, z=2 -> V^T [bh][d][s] bf16
// ---------------------------------------------------------------------------
__global__ __launch_bounds__(128) void qkv_gemm_kernel(
    const bf16_t* __restrict__ X, const bf16_t* __restrict__ WTbase,
    const float* __restrict__ bq, const float* __restrict__ bk,
    const float* __restrict__ bv,
    bf16_t* __restrict__ Qout, bf16_t* __restrict__ Kout,
    bf16_t* __restrict__ VTout) {
    const int z    = blockIdx.z;
    const int wave = threadIdx.x >> 5;
    const int lane = threadIdx.x & 31;
    const int l15  = lane & 15;
    const int half = lane >> 4;
    const int row0 = blockIdx.x * 128 + (wave >> 1) * 64;
    const int col0 = blockIdx.y * 128 + (wave & 1) * 64;

    const bf16_t* WT   = WTbase + (size_t)z * H * H;
    const float*  bias = (z == 0) ? bq : (z == 1) ? bk : bv;

    v8f c[4][4] = {};
    wave_gemm_64x64(X + (size_t)row0 * H, H, WT + (size_t)col0 * H, H, H, l15, half, c);

#pragma unroll
    for (int it = 0; it < 4; ++it)
#pragma unroll
        for (int jt = 0; jt < 4; ++jt) {
            int col = col0 + jt * 16 + l15;
            float bvv = bias[col];
            int h = col >> 6, d = col & 63;
#pragma unroll
            for (int r = 0; r < 8; ++r) {
                int row = row0 + it * 16 + half * 8 + r;      // C layout (ISA p.108)
                int b = row >> 11, s = row & (S - 1);
                float val = c[it][jt][r] + bvv;
                size_t bh = (size_t)(b * NH + h);
                if (z == 2)
                    VTout[(bh * HD + d) * S + s] = (bf16_t)val;
                else if (z == 0)
                    Qout[(bh * S + s) * HD + d] = (bf16_t)val;
                else
                    Kout[(bh * S + s) * HD + d] = (bf16_t)val;
            }
        }
}

// ---------------------------------------------------------------------------
// Flash attention.  grid 1024, block 128 (4 independent waves).
// Each wave: one 16-row query block; loops KV in blocks of 64.
// ---------------------------------------------------------------------------
__global__ __launch_bounds__(128) void attn_kernel(
    const bf16_t* __restrict__ Q, const bf16_t* __restrict__ Km,
    const bf16_t* __restrict__ Vt, const float* __restrict__ mask,
    bf16_t* __restrict__ ctxb) {
    __shared__ __bf16 lds_p[4][16][72];   // padded row stride vs bank conflicts

    const int wid  = threadIdx.x >> 5;
    const int lane = threadIdx.x & 31;
    const int l15  = lane & 15;
    const int half = lane >> 4;

    const int bid  = blockIdx.x;
    const int qgrp = bid & 31;            // 32 q-groups per (b,h)
    const int bh   = bid >> 5;            // 32 (b,h) pairs
    const int b    = bh >> 4;
    const int h    = bh & 15;
    const int q0   = (qgrp * 4 + wid) * 16;

    const bf16_t* Qb = Q  + ((size_t)bh * S + q0) * HD;
    const bf16_t* Kb = Km + (size_t)bh * S * HD;
    const bf16_t* Vb = Vt + (size_t)bh * HD * S;
    __bf16* pl = &lds_p[wid][0][0];

    // Q block as two 16x32 A-tiles (HD = 64)
    v16bf qa[2];
#pragma unroll
    for (int ks = 0; ks < 2; ++ks)
        qa[ks] = load_a_tile(Qb + (size_t)l15 * HD, ks * 32, half);

    float rmax[8], rsum[8];
#pragma unroll
    for (int r = 0; r < 8; ++r) { rmax[r] = -3.0e30f; rsum[r] = 0.0f; }
    v8f ctx[4] = {};

    for (int kv0 = 0; kv0 < S; kv0 += 64) {
        // prefetch next KV block into cache while we do softmax math
        if (kv0 + 64 < S) {
            __builtin_prefetch(Kb + (size_t)(kv0 + 64 + l15 * 4) * HD, 0, 1);
            __builtin_prefetch(Vb + (size_t)(l15 * 4) * S + kv0 + 64, 0, 1);
        }
        float pe[4][8];
        // ---- scores = Q K^T / 8 + mask (4 n-tiles of 16) ----
#pragma unroll
        for (int nt = 0; nt < 4; ++nt) {
            v8f sc = {};
#pragma unroll
            for (int ks = 0; ks < 2; ++ks) {
                v16bf kb = *(const v16bf*)(Kb + (size_t)(kv0 + nt * 16 + l15) * HD +
                                           ks * 32 + half * 16);
                sc = wmma_bf16(qa[ks], kb, sc);
            }
            float mv = mask[(size_t)b * S + kv0 + nt * 16 + l15];
#pragma unroll
            for (int r = 0; r < 8; ++r) pe[nt][r] = sc[r] * 0.125f + mv;
        }
        // ---- online softmax (row m = half*8 + r lives in this half's 16 lanes) ----
#pragma unroll
        for (int r = 0; r < 8; ++r) {
            float m = fmaxf(fmaxf(pe[0][r], pe[1][r]), fmaxf(pe[2][r], pe[3][r]));
            m = fmaxf(m, __shfl_xor(m, 1)); m = fmaxf(m, __shfl_xor(m, 2));
            m = fmaxf(m, __shfl_xor(m, 4)); m = fmaxf(m, __shfl_xor(m, 8));
            float mnew  = fmaxf(rmax[r], m);
            float alpha = __expf(rmax[r] - mnew);
            rmax[r] = mnew;
            float bs = 0.0f;
#pragma unroll
            for (int nt = 0; nt < 4; ++nt) {
                float p = __expf(pe[nt][r] - mnew);
                pe[nt][r] = p; bs += p;
            }
            bs += __shfl_xor(bs, 1); bs += __shfl_xor(bs, 2);
            bs += __shfl_xor(bs, 4); bs += __shfl_xor(bs, 8);
            rsum[r] = rsum[r] * alpha + bs;
            // rescale running context for this row
#pragma unroll
            for (int nt = 0; nt < 4; ++nt) ctx[nt][r] *= alpha;
        }
        // ---- C-layout -> A-layout relayout of P through LDS ----
#pragma unroll
        for (int nt = 0; nt < 4; ++nt)
#pragma unroll
            for (int r = 0; r < 8; ++r)
                pl[(half * 8 + r) * 72 + nt * 16 + l15] = (bf16_t)pe[nt][r];
        asm volatile("s_wait_dscnt 0" ::: "memory");
        v16bf pa[2];
#pragma unroll
        for (int ks = 0; ks < 2; ++ks)
            pa[ks] = bf_combine(
                *(const v8bf*)(pl + l15 * 72 + ks * 32 + half * 16),
                *(const v8bf*)(pl + l15 * 72 + ks * 32 + half * 16 + 8));
        // ---- ctx += P V  (V pre-transposed: [d][s]) ----
#pragma unroll
        for (int nt = 0; nt < 4; ++nt)
#pragma unroll
            for (int ks = 0; ks < 2; ++ks) {
                v16bf vb = *(const v16bf*)(Vb + (size_t)(nt * 16 + l15) * S +
                                           kv0 + ks * 32 + half * 16);
                ctx[nt] = wmma_bf16(pa[ks], vb, ctx[nt]);
            }
    }
    // ---- normalize and emit context as bf16 [token][H] ----
#pragma unroll
    for (int nt = 0; nt < 4; ++nt)
#pragma unroll
        for (int r = 0; r < 8; ++r) {
            int s = q0 + half * 8 + r;
            int d = nt * 16 + l15;
            float val = ctx[nt][r] / rsum[r];
            ctxb[((size_t)(b * S + s)) * H + h * HD + d] = (bf16_t)val;
        }
}

// ---------------------------------------------------------------------------
// Output projection + bias + residual (f32 out).  grid (32, 8), block 128.
// ---------------------------------------------------------------------------
__global__ __launch_bounds__(128) void out_gemm_kernel(
    const bf16_t* __restrict__ ctxb, const bf16_t* __restrict__ WoT,
    const float* __restrict__ bo, const float* __restrict__ resid,
    float* __restrict__ hbuf) {
    const int wave = threadIdx.x >> 5;
    const int lane = threadIdx.x & 31;
    const int l15  = lane & 15;
    const int half = lane >> 4;
    const int row0 = blockIdx.x * 128 + (wave >> 1) * 64;
    const int col0 = blockIdx.y * 128 + (wave & 1) * 64;

    v8f c[4][4] = {};
    wave_gemm_64x64(ctxb + (size_t)row0 * H, H, WoT + (size_t)col0 * H, H, H,
                    l15, half, c);
#pragma unroll
    for (int it = 0; it < 4; ++it)
#pragma unroll
        for (int jt = 0; jt < 4; ++jt) {
            int col = col0 + jt * 16 + l15;
            float bvv = bo[col];
#pragma unroll
            for (int r = 0; r < 8; ++r) {
                int row = row0 + it * 16 + half * 8 + r;
                size_t idx = (size_t)row * H + col;
                hbuf[idx] = c[it][jt][r] + bvv + resid[idx];
            }
        }
}

// ---------------------------------------------------------------------------
// LayerNorm over H=1024.  grid 4096, block 256 (4 floats per thread).
// ---------------------------------------------------------------------------
__global__ __launch_bounds__(256) void ln_kernel(
    const float* __restrict__ hbuf, const float* __restrict__ g,
    const float* __restrict__ be, float* __restrict__ out) {
    const int row = blockIdx.x;
    const int t   = threadIdx.x;
    __shared__ float sm[8];

    float4 v = ((const float4*)(hbuf + (size_t)row * H))[t];
    float s = v.x + v.y + v.z + v.w;
#pragma unroll
    for (int off = 16; off > 0; off >>= 1) s += __shfl_xor(s, off);
    if ((t & 31) == 0) sm[t >> 5] = s;
    __syncthreads();
    float mean = 0.0f;
#pragma unroll
    for (int i = 0; i < 8; ++i) mean += sm[i];
    mean *= (1.0f / H);
    __syncthreads();

    float dx0 = v.x - mean, dx1 = v.y - mean, dx2 = v.z - mean, dx3 = v.w - mean;
    float q = dx0 * dx0 + dx1 * dx1 + dx2 * dx2 + dx3 * dx3;
#pragma unroll
    for (int off = 16; off > 0; off >>= 1) q += __shfl_xor(q, off);
    if ((t & 31) == 0) sm[t >> 5] = q;
    __syncthreads();
    float var = 0.0f;
#pragma unroll
    for (int i = 0; i < 8; ++i) var += sm[i];
    var *= (1.0f / H);
    float inv = rsqrtf(var + 1e-12f);

    float4 gv = ((const float4*)g)[t];
    float4 bv = ((const float4*)be)[t];
    float4 o;
    o.x = gv.x * dx0 * inv + bv.x;
    o.y = gv.y * dx1 * inv + bv.y;
    o.z = gv.z * dx2 * inv + bv.z;
    o.w = gv.w * dx3 * inv + bv.w;
    ((float4*)(out + (size_t)row * H))[t] = o;
}

// ---------------------------------------------------------------------------
// Host orchestration
// ---------------------------------------------------------------------------
extern "C" void kernel_launch(void* const* d_in, const int* in_sizes, int n_in,
                              void* d_out, int out_size, void* d_ws, size_t ws_size,
                              hipStream_t stream) {
    const float* hidden = (const float*)d_in[0];
    const float* mask   = (const float*)d_in[1];
    const float* Wq = (const float*)d_in[2];  const float* bq = (const float*)d_in[3];
    const float* Wk = (const float*)d_in[4];  const float* bk = (const float*)d_in[5];
    const float* Wv = (const float*)d_in[6];  const float* bv = (const float*)d_in[7];
    const float* Wo = (const float*)d_in[8];  const float* bo = (const float*)d_in[9];
    const float* gamma = (const float*)d_in[10];
    const float* beta  = (const float*)d_in[11];
    float* out = (float*)d_out;

    char* ws = (char*)d_ws;
    bf16_t* hsb  = (bf16_t*)(ws);                        //  8 MB hidden bf16
    bf16_t* WTb  = (bf16_t*)(ws + (size_t)( 8 << 20));   //  8 MB WqT|WkT|WvT|WoT
    bf16_t* Qb   = (bf16_t*)(ws + (size_t)(16 << 20));   //  8 MB
    bf16_t* Kb   = (bf16_t*)(ws + (size_t)(24 << 20));   //  8 MB
    bf16_t* Vtb  = (bf16_t*)(ws + (size_t)(32 << 20));   //  8 MB
    bf16_t* ctxb = (bf16_t*)(ws + (size_t)(40 << 20));   //  8 MB
    float*  hbuf = (float*) (ws + (size_t)(48 << 20));   // 16 MB

    const int nelt = NTOK * H;
    f32_to_bf16_kernel<<<nelt / 256, 256, 0, stream>>>(hidden, hsb, nelt);

    dim3 tgrid(H / 16, H / 16);
    transpose_bf16_kernel<<<tgrid, 256, 0, stream>>>(Wq, WTb + 0 * (size_t)H * H);
    transpose_bf16_kernel<<<tgrid, 256, 0, stream>>>(Wk, WTb + 1 * (size_t)H * H);
    transpose_bf16_kernel<<<tgrid, 256, 0, stream>>>(Wv, WTb + 2 * (size_t)H * H);
    transpose_bf16_kernel<<<tgrid, 256, 0, stream>>>(Wo, WTb + 3 * (size_t)H * H);

    qkv_gemm_kernel<<<dim3(NTOK / 128, H / 128, 3), 128, 0, stream>>>(
        hsb, WTb, bq, bk, bv, Qb, Kb, Vtb);

    attn_kernel<<<1024, 128, 0, stream>>>(Qb, Kb, Vtb, mask, ctxb);

    out_gemm_kernel<<<dim3(NTOK / 128, H / 128), 128, 0, stream>>>(
        ctxb, WTb + 3 * (size_t)H * H, bo, hidden, hbuf);

    ln_kernel<<<NTOK, 256, 0, stream>>>(hbuf, gamma, beta, out);
}
